// GlobalAttention_65137474011747
// MI455X (gfx1250) — compile-verified, matrix operand order checked by reading, not run on hardware
//
#include <hip/hip_runtime.h>
#include <hip/hip_bf16.h>

#define RR   256   // rows
#define SS   1024  // sequence
#define CIN  256   // input channels
#define CH   32    // head dim
#define NH   8     // heads

typedef __attribute__((ext_vector_type(16))) __bf16 v16bf;
typedef __attribute__((ext_vector_type(8)))  float  v8f;

// ---- WMMA helpers -----------------------------------------------------------

__device__ __forceinline__ v8f wmma_bf16(v16bf a, v16bf b, v8f c) {
    // D = A(16x32 bf16) x B(32x16 bf16) + C(16x16 f32)
    return __builtin_amdgcn_wmma_f32_16x16x32_bf16(
        /*neg_a=*/false, a, /*neg_b=*/false, b,
        /*c_mod=*/(short)0, c, /*reuse_a=*/false, /*reuse_b=*/false);
}

// Load a 16x32 bf16 A-fragment (or the symmetric 32x16 B-fragment from an
// N-major [N][K] tile) from LDS. ISA layout: lane M = l&15; kbase = (l&16)?8:0;
// VGPR v holds K = kbase + (v&3)*2 + ((v&4)?16:0) (+0/+1 in lo/hi halves).
__device__ __forceinline__ v16bf load_frag_lds(const __bf16* base, int rstride,
                                               int kk, int lane) {
    union { v16bf v; unsigned u[8]; } f;
    const unsigned* p =
        (const unsigned*)(base + (size_t)(lane & 15) * rstride + kk);
    const int kbase = (lane & 16) ? 8 : 0;
#pragma unroll
    for (int vg = 0; vg < 8; ++vg) {
        int off = kbase + ((vg & 3) << 1) + ((vg & 4) ? 16 : 0);
        f.u[vg] = p[off >> 1];
    }
    return f.v;
}

// Build a B-fragment from a K-major global f32 matrix W[k][n] with row stride
// `nstride` (weights live in L2; converted to bf16 on the fly).
__device__ __forceinline__ v16bf load_frag_global_kmajor(const float* W, int nstride,
                                                         int kk, int ncol, int lane) {
    union { v16bf v; __bf16 h[16]; } f;
    const int kbase = kk + ((lane & 16) ? 8 : 0);
#pragma unroll
    for (int vg = 0; vg < 8; ++vg) {
        int k0 = kbase + ((vg & 3) << 1) + ((vg & 4) ? 16 : 0);
        f.h[2 * vg]     = (__bf16)W[(size_t)k0 * nstride + ncol];
        f.h[2 * vg + 1] = (__bf16)W[(size_t)(k0 + 1) * nstride + ncol];
    }
    return f.v;
}

// ---- vectorized f32 -> bf16 tile staging ------------------------------------

__device__ __forceinline__ unsigned pack_bf16x2(float a, float b) {
    union { __bf16 h[2]; unsigned u; } p;
    p.h[0] = (__bf16)a;
    p.h[1] = (__bf16)b;
    return p.u;
}

// Copy nfloat4*4 f32 elements from global to a contiguous bf16 LDS array using
// global_load_b128 + ds_store_b64 per element-quad.
__device__ __forceinline__ void stage_tile_bf16(__bf16* dst, const float* __restrict__ src,
                                                int nfloat4, int tid, int nthreads) {
    const float4* s4 = (const float4*)src;
    uint2*        d2 = (uint2*)dst;
    for (int i = tid; i < nfloat4; i += nthreads) {
        float4 v = s4[i];
        uint2  o;
        o.x = pack_bf16x2(v.x, v.y);
        o.y = pack_bf16x2(v.z, v.w);
        d2[i] = o;
    }
}

// ---- Kernel 1: masked mean-pool + q projection + flash attention per row ----
// grid = RR blocks, 256 threads (8 waves). wave h owns head h; lane = channel.
__global__ void __launch_bounds__(256)
attn_kernel(const float* __restrict__ m, const float* __restrict__ mask,
            const float* __restrict__ Wq, const float* __restrict__ Wk,
            const float* __restrict__ Wv, float* __restrict__ o_ws) {
    __shared__ float  q_ld[CIN];
    __shared__ float  qh[CIN];                  // (q@Wq) * c^-0.5, n = h*32+c
    __shared__ __bf16 WkT[CH][CIN];             // Wk transposed -> N-major
    __shared__ __bf16 WvT[CH][CIN];
    __shared__ __align__(16) __bf16 mtile[64][CIN];  // 64-token bf16 stage
    __shared__ float  ktile[64][CH];
    __shared__ float  vtile[64][CH];
    __shared__ float  p_lds[NH][64];
    __shared__ float4 qpart[4][64];
    __shared__ float  mpart[4][64];

    const int r    = blockIdx.x;
    const int tid  = threadIdx.x;
    const int lane = tid & 31;
    const int wave = tid >> 5;

    const float* mrow    = m    + (size_t)r * SS * CIN;
    const float* maskrow = mask + (size_t)r * SS;

    // stage WkT / WvT (coalesced read, LDS transpose write)
    for (int i = tid; i < CH * CIN; i += 256) {
        int c = i >> 5, j = i & 31;
        WkT[j][c] = (__bf16)Wk[c * CH + j];
        WvT[j][c] = (__bf16)Wv[c * CH + j];
    }

    // masked mean pool with b128 loads: 64 channel-quads x 4 seq-segments
    {
        const int cq  = (tid & 63);        // channel quad index (channels 4cq..4cq+3)
        const int seg = tid >> 6;          // sequence segment 0..3
        float4 acc = {0.f, 0.f, 0.f, 0.f};
        float  msum = 0.f;
        const int sbeg = seg * (SS / 4), send = sbeg + (SS / 4);
        for (int s = sbeg; s < send; ++s) {
            float  w  = maskrow[s];
            float4 mv = *(const float4*)&mrow[(size_t)s * CIN + cq * 4];
            acc.x += mv.x * w; acc.y += mv.y * w;
            acc.z += mv.z * w; acc.w += mv.w * w;
            msum  += w;
        }
        qpart[seg][cq] = acc;
        mpart[seg][cq] = msum;
    }
    __syncthreads();
    if (tid < 64) {
        float4 a0 = qpart[0][tid], a1 = qpart[1][tid],
               a2 = qpart[2][tid], a3 = qpart[3][tid];
        float  mt = mpart[0][tid] + mpart[1][tid] + mpart[2][tid] + mpart[3][tid];
        float  inv = 1.0f / (mt + 1e-10f);
        q_ld[4 * tid + 0] = (a0.x + a1.x + a2.x + a3.x) * inv;
        q_ld[4 * tid + 1] = (a0.y + a1.y + a2.y + a3.y) * inv;
        q_ld[4 * tid + 2] = (a0.z + a1.z + a2.z + a3.z) * inv;
        q_ld[4 * tid + 3] = (a0.w + a1.w + a2.w + a3.w) * inv;
    }
    __syncthreads();

    // qh = (q @ Wq) * CH^-0.5   (Wq: [CIN][NH*CH] row-major)
    {
        float a2 = 0.f;
        for (int c = 0; c < CIN; ++c) a2 += q_ld[c] * Wq[c * (NH * CH) + tid];
        qh[tid] = a2 * 0.17677669529663687f;   // 1/sqrt(32)
    }
    __syncthreads();

    // online-softmax state for head `wave`, channel `lane`
    float run_max = -3.0e38f, Lsum = 0.f, Oacc = 0.f;

    const int mt = wave & 3;   // token 16-tile within 64
    const int nt = wave >> 2;  // channel 16-tile within 32

    for (int st = 0; st < SS / 64; ++st) {
        const int s0 = st * 64;
        stage_tile_bf16(&mtile[0][0], mrow + (size_t)s0 * CIN, 64 * CIN / 4, tid, 256);
        // prefetch next tile while this one is being consumed
        if (st + 1 < SS / 64)
            __builtin_prefetch(mrow + (size_t)(s0 + 64) * CIN + tid * 64, 0, 0);
        __syncthreads();

        // k / v GEMM: one 16x16 D tile per wave for each, A-fragment shared
        {
            v8f ak = {0.f, 0.f, 0.f, 0.f, 0.f, 0.f, 0.f, 0.f};
            v8f av = {0.f, 0.f, 0.f, 0.f, 0.f, 0.f, 0.f, 0.f};
#pragma unroll
            for (int kk = 0; kk < CIN; kk += 32) {
                v16bf afrag = load_frag_lds(&mtile[mt * 16][0], CIN, kk, lane);
                v16bf bk    = load_frag_lds(&WkT[nt * 16][0],   CIN, kk, lane);
                v16bf bv    = load_frag_lds(&WvT[nt * 16][0],   CIN, kk, lane);
                ak = wmma_bf16(afrag, bk, ak);
                av = wmma_bf16(afrag, bv, av);
            }
            const int trowb = mt * 16 + ((lane & 16) ? 8 : 0);
            const int col   = nt * 16 + (lane & 15);
#pragma unroll
            for (int j = 0; j < 8; ++j) {
                ktile[trowb + j][col] = ak[j];
                vtile[trowb + j][col] = av[j];
            }
        }
        __syncthreads();

        // scoring + online softmax: wave = head, 2 tokens per lane
        const int h  = wave;
        const int t0 = lane, t1 = lane + 32;
        float sc0 = 0.f, sc1 = 0.f;
#pragma unroll
        for (int j = 0; j < CH; ++j) {
            float qv = qh[h * CH + j];
            sc0 += qv * ktile[t0][j];
            sc1 += qv * ktile[t1][j];
        }
        sc0 += 1.0e9f * (maskrow[s0 + t0] - 1.0f);
        sc1 += 1.0e9f * (maskrow[s0 + t1] - 1.0f);

        float tmax = fmaxf(sc0, sc1);
#pragma unroll
        for (int d = 16; d; d >>= 1) tmax = fmaxf(tmax, __shfl_xor(tmax, d, 32));
        float nmax  = fmaxf(run_max, tmax);
        float alpha = __expf(run_max - nmax);
        run_max = nmax;

        float p0 = __expf(sc0 - nmax), p1 = __expf(sc1 - nmax);
        float ps = p0 + p1;
#pragma unroll
        for (int d = 16; d; d >>= 1) ps += __shfl_xor(ps, d, 32);
        Lsum = Lsum * alpha + ps;

        p_lds[h][t0] = p0;
        p_lds[h][t1] = p1;
        __syncthreads();

        Oacc *= alpha;
        for (int t = 0; t < 64; ++t) Oacc += p_lds[h][t] * vtile[t][lane];
        __syncthreads();
    }

    o_ws[r * (NH * CH) + wave * CH + lane] = Oacc / Lsum;  // o[r, h*32+c]
}

// ---- Kernel 2: gating + output projection -----------------------------------
// grid = (SS/64, RR), 256 threads. Each wave owns 2 N-panels; weight B-frags
// are held in registers across the 4 M-tiles (weights read once per block).
__global__ void __launch_bounds__(256)
out_kernel(const float* __restrict__ m,  const float* __restrict__ Wg,
           const float* __restrict__ bg, const float* __restrict__ Wo,
           const float* __restrict__ bo, const float* __restrict__ o_ws,
           float* __restrict__ out) {
    __shared__ __align__(16) __bf16 mtile[64][CIN];
    __shared__ __align__(16) __bf16 utile[64][CIN];  // sigmoid(m@Wg+bg)*o (bf16)
    __shared__ float o_row[CIN];

    const int r    = blockIdx.y;
    const int st   = blockIdx.x;
    const int tid  = threadIdx.x;
    const int lane = tid & 31;
    const int wave = tid >> 5;
    const int s0   = st * 64;

    const float* mrow = m + ((size_t)r * SS + s0) * CIN;

    // async copy of o_row (f32, 1 KB) into LDS, overlapped with m-tile staging
    {
        unsigned     ldsoff = (unsigned)(uintptr_t)(&o_row[tid]);
        unsigned     voff   = (unsigned)(tid * 4);
        const float* srow   = o_ws + (size_t)r * CIN;
        asm volatile("global_load_async_to_lds_b32 %0, %1, %2"
                     :: "v"(ldsoff), "v"(voff), "s"(srow) : "memory");
    }
    stage_tile_bf16(&mtile[0][0], mrow, 64 * CIN / 4, tid, 256);
    asm volatile("s_wait_asynccnt 0x0" ::: "memory");
    __syncthreads();

    // ---- g stage: u = sigmoid(m@Wg + bg) * o_row ----
#pragma unroll
    for (int nn = 0; nn < 2; ++nn) {
        const int nt   = wave * 2 + nn;
        const int ncol = nt * 16 + (lane & 15);
        v16bf bfr[8];
#pragma unroll
        for (int k8 = 0; k8 < 8; ++k8)
            bfr[k8] = load_frag_global_kmajor(Wg, NH * CH, k8 * 32, ncol, lane);
        const float bgv = bg[ncol];
        const float ogv = o_row[ncol];
#pragma unroll
        for (int mt2 = 0; mt2 < 4; ++mt2) {
            v8f acc = {0.f, 0.f, 0.f, 0.f, 0.f, 0.f, 0.f, 0.f};
#pragma unroll
            for (int k8 = 0; k8 < 8; ++k8) {
                v16bf afrag = load_frag_lds(&mtile[mt2 * 16][0], CIN, k8 * 32, lane);
                acc = wmma_bf16(afrag, bfr[k8], acc);
            }
            const int trowb = mt2 * 16 + ((lane & 16) ? 8 : 0);
#pragma unroll
            for (int j = 0; j < 8; ++j) {
                float x    = acc[j] + bgv;
                float gsig = 1.0f / (1.0f + __expf(-x));
                utile[trowb + j][ncol] = (__bf16)(gsig * ogv);
            }
        }
    }
    __syncthreads();

    // ---- output stage: out = u @ Wo + bo ----
    float* orow = out + ((size_t)r * SS + s0) * CIN;
#pragma unroll
    for (int nn = 0; nn < 2; ++nn) {
        const int nt   = wave * 2 + nn;
        const int ncol = nt * 16 + (lane & 15);
        v16bf bfr[8];
#pragma unroll
        for (int k8 = 0; k8 < 8; ++k8)
            bfr[k8] = load_frag_global_kmajor(Wo, CIN, k8 * 32, ncol, lane);
        const float bov = bo[ncol];
#pragma unroll
        for (int mt2 = 0; mt2 < 4; ++mt2) {
            v8f acc = {0.f, 0.f, 0.f, 0.f, 0.f, 0.f, 0.f, 0.f};
#pragma unroll
            for (int k8 = 0; k8 < 8; ++k8) {
                v16bf afrag = load_frag_lds(&utile[mt2 * 16][0], CIN, k8 * 32, lane);
                acc = wmma_bf16(afrag, bfr[k8], acc);
            }
            const int trowb = mt2 * 16 + ((lane & 16) ? 8 : 0);
#pragma unroll
            for (int j = 0; j < 8; ++j)
                orow[(size_t)(trowb + j) * CIN + ncol] = acc[j] + bov;
        }
    }
}

// ---- host entry -------------------------------------------------------------
extern "C" void kernel_launch(void* const* d_in, const int* in_sizes, int n_in,
                              void* d_out, int out_size, void* d_ws, size_t ws_size,
                              hipStream_t stream) {
    const float* m    = (const float*)d_in[0];
    const float* mask = (const float*)d_in[1];
    const float* Wq   = (const float*)d_in[2];
    const float* Wk   = (const float*)d_in[3];
    const float* Wv   = (const float*)d_in[4];
    const float* Wg   = (const float*)d_in[5];
    const float* bg   = (const float*)d_in[6];
    const float* Wo   = (const float*)d_in[7];
    const float* bo   = (const float*)d_in[8];
    float* out  = (float*)d_out;
    float* o_ws = (float*)d_ws;   // RR * NH*CH f32 = 256 KB

    attn_kernel<<<RR, 256, 0, stream>>>(m, mask, Wq, Wk, Wv, o_ws);
    out_kernel<<<dim3(SS / 64, RR), 256, 0, stream>>>(m, Wg, bg, Wo, bo, o_ws, out);
}
